// SelfSupervisedCombinedLoss_34746285424745
// MI455X (gfx1250) — compile-verified, matrix operand order checked by reading, not run on hardware
//
#include <hip/hip_runtime.h>

// CDNA5 / gfx1250. Wave32. WMMA types per CDNA5_HIP.md.
typedef __attribute__((ext_vector_type(16))) _Float16 v16h;
typedef __attribute__((ext_vector_type(8)))  float    v8f;

#define IMG   512
#define BLK   1024          // 32 wave32 waves
#define NIMG  64
#define NACC  8             // 0:n2v_num 1:mask_cnt 2:S1 3:S2 4:S3 5:tv_v 6:tv_h 7:pad

__device__ __forceinline__ float clip01(float x) { return fminf(fmaxf(x, 0.0f), 1.0f); }

__device__ __forceinline__ float4 clip4(float4 v) {
    v.x = clip01(v.x); v.y = clip01(v.y); v.z = clip01(v.z); v.w = clip01(v.w);
    return v;
}

// One 2x2 Haar block: accumulate sum(min(|detail|,thr)) over ch/cv/cd, return LL.
__device__ __forceinline__ float haar_acc(float a, float b, float c, float d,
                                          float thr, float& S) {
    const float ll = 0.5f * (a + b + c + d);
    const float ch = 0.5f * (a + b - c - d);
    const float cv = 0.5f * (a - b + c - d);
    const float cd = 0.5f * (a - b - c + d);
    S += fminf(fabsf(ch), thr) + fminf(fabsf(cv), thr) + fminf(fabsf(cd), thr);
    return ll;
}

__global__ void zero_ws(float* ws) {
    if (threadIdx.x < NACC) ws[threadIdx.x] = 0.0f;
}

// One workgroup per image. Dynamic LDS: ll1 (256*256 f32 = 256KB) + ll2 (128*128 = 64KB)
// = 320KB, the full CDNA5 per-WGP LDS — the whole wavelet pyramid stays on-chip,
// so only level-1 touches HBM. Single fused streaming pass over pred/noisy/mask.
__global__ __launch_bounds__(BLK)
void loss_main(const float* __restrict__ pred,
               const float* __restrict__ noisy,
               const unsigned char* __restrict__ mask,
               float* __restrict__ ws)
{
    extern __shared__ float lds[];
    float* ll1 = lds;                 // 65536 floats
    float* ll2 = lds + 256 * 256;     // 16384 floats

    const int    tid  = threadIdx.x;
    const size_t base = (size_t)blockIdx.x * IMG * IMG;

    const float thr1 = 50.0f / 4.0f / 255.0f;   // finest details (level_idx=3)
    const float thr2 = 50.0f / 2.0f / 255.0f;
    const float thr3 = 50.0f / 255.0f;          // coarsest  (level_idx=1)

    float a_n2v = 0.f, a_cnt = 0.f, a_s1 = 0.f, a_s2 = 0.f, a_s3 = 0.f;
    float a_tvv = 0.f, a_tvh = 0.f;

    // ---- Phase 1: stream 512x512 image; fuse clip + N2V + TV + level-1 Haar ----
    // Task = one output row r (of 256) x one column-pair cp (of 128):
    // reads input rows 2r,2r+1, cols 4cp..4cp+3 as float4 (coalesced B128 loads).
    for (int task = tid; task < 256 * 128; task += BLK) {
        const int r   = task >> 7;
        const int cp  = task & 127;
        const int col = cp << 2;

        const float* p0 = pred + base + (size_t)(2 * r) * IMG + col;
        const float* p1 = p0 + IMG;
        __builtin_prefetch(p0 + 16 * IMG, 0, 1);   // global_prefetch_b8 (next iter rows)

        const float4 t0 = clip4(*(const float4*)p0);
        const float4 t1 = clip4(*(const float4*)p1);

        // N2V: |p - noisy| * mask (mask is 1 byte/elem)
        {
            const float4 n0 = *(const float4*)(noisy + base + (size_t)(2 * r) * IMG + col);
            const float4 n1 = *(const float4*)(noisy + base + (size_t)(2 * r + 1) * IMG + col);
            const unsigned m0 = *(const unsigned*)(mask + base + (size_t)(2 * r) * IMG + col);
            const unsigned m1 = *(const unsigned*)(mask + base + (size_t)(2 * r + 1) * IMG + col);
            const float b00 = (m0 & 0xFFu) ? 1.f : 0.f, b01 = (m0 & 0xFF00u) ? 1.f : 0.f;
            const float b02 = (m0 & 0xFF0000u) ? 1.f : 0.f, b03 = (m0 & 0xFF000000u) ? 1.f : 0.f;
            const float b10 = (m1 & 0xFFu) ? 1.f : 0.f, b11 = (m1 & 0xFF00u) ? 1.f : 0.f;
            const float b12 = (m1 & 0xFF0000u) ? 1.f : 0.f, b13 = (m1 & 0xFF000000u) ? 1.f : 0.f;
            a_n2v += fabsf(t0.x - n0.x) * b00 + fabsf(t0.y - n0.y) * b01
                   + fabsf(t0.z - n0.z) * b02 + fabsf(t0.w - n0.w) * b03
                   + fabsf(t1.x - n1.x) * b10 + fabsf(t1.y - n1.y) * b11
                   + fabsf(t1.z - n1.z) * b12 + fabsf(t1.w - n1.w) * b13;
            a_cnt += b00 + b01 + b02 + b03 + b10 + b11 + b12 + b13;
        }

        // TV horizontal: 3 in-block pairs per row + cross-block seam (cols 4cp+3,4cp+4)
        a_tvh += fabsf(t0.y - t0.x) + fabsf(t0.z - t0.y) + fabsf(t0.w - t0.z)
               + fabsf(t1.y - t1.x) + fabsf(t1.z - t1.y) + fabsf(t1.w - t1.z);
        if (col + 4 < IMG) {
            a_tvh += fabsf(clip01(p0[4]) - t0.w) + fabsf(clip01(p1[4]) - t1.w);
        }
        // TV vertical: pair (2r,2r+1) local; pair (2r+1,2r+2) via one extra row load
        a_tvv += fabsf(t1.x - t0.x) + fabsf(t1.y - t0.y)
               + fabsf(t1.z - t0.z) + fabsf(t1.w - t0.w);
        if (2 * r + 2 < IMG) {
            const float4 u = clip4(*(const float4*)(p1 + IMG));
            a_tvv += fabsf(u.x - t1.x) + fabsf(u.y - t1.y)
                   + fabsf(u.z - t1.z) + fabsf(u.w - t1.w);
        }

        // Level-1 Haar: two 2x2 blocks -> two LL outputs into LDS
        ll1[r * 256 + 2 * cp]     = haar_acc(t0.x, t0.y, t1.x, t1.y, thr1, a_s1);
        ll1[r * 256 + 2 * cp + 1] = haar_acc(t0.z, t0.w, t1.z, t1.w, thr1, a_s1);
    }
    __syncthreads();

    // ---- Phase 2: 256x256 LL1 (LDS) -> 128x128 LL2 (LDS), ds_load_b128 reads ----
    for (int task = tid; task < 128 * 64; task += BLK) {
        const int r = task >> 6, cp = task & 63, col = cp << 2;
        const float4 t0 = *(const float4*)(ll1 + (2 * r) * 256 + col);
        const float4 t1 = *(const float4*)(ll1 + (2 * r + 1) * 256 + col);
        ll2[r * 128 + 2 * cp]     = haar_acc(t0.x, t0.y, t1.x, t1.y, thr2, a_s2);
        ll2[r * 128 + 2 * cp + 1] = haar_acc(t0.z, t0.w, t1.z, t1.w, thr2, a_s2);
    }
    __syncthreads();

    // ---- Phase 3: 128x128 LL2 -> level-3 detail sums only ----
    for (int task = tid; task < 64 * 32; task += BLK) {
        const int r = task >> 5, cp = task & 31, col = cp << 2;
        const float4 t0 = *(const float4*)(ll2 + (2 * r) * 128 + col);
        const float4 t1 = *(const float4*)(ll2 + (2 * r + 1) * 128 + col);
        (void)haar_acc(t0.x, t0.y, t1.x, t1.y, thr3, a_s3);
        (void)haar_acc(t0.z, t0.w, t1.z, t1.w, thr3, a_s3);
    }

    // ---- Wave32 shuffle reduction + fp32 global atomics ----
    float vals[7] = { a_n2v, a_cnt, a_s1, a_s2, a_s3, a_tvv, a_tvh };
#pragma unroll
    for (int k = 0; k < 7; ++k) {
        float v = vals[k];
#pragma unroll
        for (int off = 16; off > 0; off >>= 1) v += __shfl_down(v, off, 32);
        if ((tid & 31) == 0) atomicAdd(&ws[k], v);   // global_atomic_add_f32
    }
}

// Final weighted combination as a tensor-op dot product:
// A row 0 holds the 8 weighted terms (K=0..7), B col 0 holds ones,
// so D[0][0] = sum(terms) via v_wmma_f32_16x16x32_f16. Single wave, EXEC all-1s.
__global__ void finalize_wmma(const float* __restrict__ ws, float* __restrict__ out)
{
    const int lane = threadIdx.x;   // 0..31

    const float n2v_num = ws[0];
    const float cnt     = fmaxf(ws[1], 1.0f);
    // wav = WAV_W * [ (1/3)*S1/(3*64*256^2) + (1/2)*S2/(3*64*128^2) + S3/(3*64*64^2) ]
    const float w1 = 0.2f * (1.0f / 3.0f) / (3.0f * 64.0f * 256.0f * 256.0f);
    const float w2 = 0.2f * 0.5f          / (3.0f * 64.0f * 128.0f * 128.0f);
    const float w3 = 0.2f                 / (3.0f * 64.0f * 64.0f  * 64.0f);
    const float wv = 0.01f / (64.0f * 511.0f * 512.0f);   // TV vertical mean
    const float wh = 0.01f / (64.0f * 512.0f * 511.0f);   // TV horizontal mean

    float t[8];
    t[0] = 1.0f * n2v_num / cnt;   // N2V_W = 1.0
    t[1] = ws[2] * w1;
    t[2] = ws[3] * w2;
    t[3] = ws[4] * w3;
    t[4] = ws[5] * wv;
    t[5] = ws[6] * wh;
    t[6] = 0.0f;
    t[7] = 0.0f;

    const float sel = (lane == 0) ? 1.0f : 0.0f;   // selection by multiply: EXEC stays all-1s
    v16h a, b;
#pragma unroll
    for (int i = 0; i < 16; ++i) { a[i] = (_Float16)0.0f; b[i] = (_Float16)0.0f; }
#pragma unroll
    for (int i = 0; i < 8; ++i) {
        a[i] = (_Float16)(t[i] * sel);   // A[M=0][K=i] on lane 0
        b[i] = (_Float16)sel;            // B[K=i][N=0] on lane 0
    }

    v8f c = {};
    c = __builtin_amdgcn_wmma_f32_16x16x32_f16(
            /*neg_a=*/false, a, /*neg_b=*/false, b,
            /*c_mod=*/(short)0, c, /*reuse_a=*/false, /*reuse_b=*/false);

    if (lane == 0) out[0] = c[0];   // D[0][0]
}

extern "C" void kernel_launch(void* const* d_in, const int* in_sizes, int n_in,
                              void* d_out, int out_size, void* d_ws, size_t ws_size,
                              hipStream_t stream) {
    (void)in_sizes; (void)n_in; (void)out_size; (void)ws_size;
    const float*         pred  = (const float*)d_in[0];
    const float*         noisy = (const float*)d_in[1];
    const unsigned char* mask  = (const unsigned char*)d_in[2];  // jax bool -> 1 byte
    float* out = (float*)d_out;
    float* ws  = (float*)d_ws;

    zero_ws<<<1, 32, 0, stream>>>(ws);
    // 320 KB dynamic LDS per workgroup: ll1 (256KB) + ll2 (64KB)
    loss_main<<<NIMG, BLK, 320 * 1024, stream>>>(pred, noisy, mask, ws);
    finalize_wmma<<<1, 32, 0, stream>>>(ws, out);
}
